// MultiHeadAttention_20401094656078
// MI455X (gfx1250) — compile-verified, matrix operand order checked by reading, not run on hardware
//
#include <hip/hip_runtime.h>

#define D_MODEL 1024
#define SEQ 2048
#define BATCH 4
#define NHEAD 16
#define DK 64
#define ATT_WAVES 4

typedef __attribute__((ext_vector_type(16))) __bf16 v16bf;
typedef __attribute__((ext_vector_type(8)))  float  v8f;

union Frag {
    v16bf v;
    uint4 u[2];
    __bf16 h[16];
};

__device__ __forceinline__ v8f zero8() {
    v8f z;
#pragma unroll
    for (int i = 0; i < 8; ++i) z[i] = 0.0f;
    return z;
}

__device__ __forceinline__ v8f wmma_bf16(v16bf a, v16bf b, v8f c) {
    // D = A(16x32 bf16) * B(32x16 bf16) + C(16x16 f32)
    return __builtin_amdgcn_wmma_f32_16x16x32_bf16(
        /*neg_a=*/false, a, /*neg_b=*/false, b,
        /*c_mod=*/(short)0, c, /*reuse_a=*/false, /*reuse_b=*/false);
}

__device__ __forceinline__ void wave_lds_sync() {
    // Same-wave cross-lane LDS dependency: DS ops are in-order per wave;
    // pin the compiler and drain DScnt.
    __builtin_amdgcn_wave_barrier();
    asm volatile("s_wait_dscnt 0" ::: "memory");
    __builtin_amdgcn_wave_barrier();
}

// per-lane async copy of 16 bytes global -> LDS (ASYNCcnt tracked, no VGPR data)
__device__ __forceinline__ void async_g2l_16B(unsigned lds_off, const __bf16* gaddr) {
    asm volatile("global_load_async_to_lds_b128 %0, %1, off"
                 :: "v"(lds_off), "v"(gaddr) : "memory");
}

__device__ __forceinline__ void wait_async0() {
    asm volatile("s_wait_asynccnt 0" ::: "memory");
    __builtin_amdgcn_wave_barrier();
}

// ---------------------------------------------------------------------------
// Kernel 1: fp32 -> bf16 weight cast
// ---------------------------------------------------------------------------
__global__ void cast_w_kernel(const float* __restrict__ src,
                              __bf16* __restrict__ dst, int n) {
    int idx = (blockIdx.x * blockDim.x + threadIdx.x) * 4;
    if (idx >= n) return;
    float4 v = *(const float4*)(src + idx);
    union { __bf16 h[4]; uint2 u; } o;
    o.h[0] = (__bf16)v.x; o.h[1] = (__bf16)v.y;
    o.h[2] = (__bf16)v.z; o.h[3] = (__bf16)v.w;
    *(uint2*)(dst + idx) = o.u;
}

// ---------------------------------------------------------------------------
// Kernel 2: q/k/v projections  O = X * W^T + b   (X fp32, W bf16, O bf16)
// wave -> 32x64 output tile (2 M-subtiles x 4 N-subtiles, B frags reused x2)
// ---------------------------------------------------------------------------
__global__ __launch_bounds__(256) void proj_qkv_kernel(
    const float* __restrict__ Q, const float* __restrict__ K,
    const float* __restrict__ V,
    const __bf16* __restrict__ Wq, const __bf16* __restrict__ Wk,
    const __bf16* __restrict__ Wv,
    const float* __restrict__ bq, const float* __restrict__ bk,
    const float* __restrict__ bv,
    __bf16* __restrict__ qp, __bf16* __restrict__ kp, __bf16* __restrict__ vp)
{
    const int z = blockIdx.z;
    const float*  X    = (z == 0) ? Q  : (z == 1) ? K  : V;
    const __bf16* W    = (z == 0) ? Wq : (z == 1) ? Wk : Wv;
    const float*  bias = (z == 0) ? bq : (z == 1) ? bk : bv;
    __bf16*       O    = (z == 0) ? qp : (z == 1) ? kp : vp;

    const int wave = threadIdx.x >> 5;
    const int lane = threadIdx.x & 31;
    const int row  = lane & 15;
    const int kg   = (lane >> 4) << 3;          // 0 or 8 (K-group / row-group)
    const int m0   = blockIdx.x * 32;
    const int n0   = (blockIdx.y * 8 + wave) * 64;

    v8f acc[2][4];
#pragma unroll
    for (int mi = 0; mi < 2; ++mi)
#pragma unroll
        for (int nt = 0; nt < 4; ++nt) acc[mi][nt] = zero8();

    const float* xrow0 = X + (size_t)(m0 + row) * D_MODEL + kg;
    const float* xrow1 = xrow0 + (size_t)16 * D_MODEL;

    for (int k0 = 0; k0 < D_MODEL; k0 += 32) {
        Frag a[2];
#pragma unroll
        for (int mi = 0; mi < 2; ++mi) {
            const float* xr = (mi == 0) ? xrow0 : xrow1;
            float4 f0 = *(const float4*)(xr + k0);
            float4 f1 = *(const float4*)(xr + k0 + 4);
            float4 f2 = *(const float4*)(xr + k0 + 16);
            float4 f3 = *(const float4*)(xr + k0 + 20);
            a[mi].h[0]  = (__bf16)f0.x; a[mi].h[1]  = (__bf16)f0.y;
            a[mi].h[2]  = (__bf16)f0.z; a[mi].h[3]  = (__bf16)f0.w;
            a[mi].h[4]  = (__bf16)f1.x; a[mi].h[5]  = (__bf16)f1.y;
            a[mi].h[6]  = (__bf16)f1.z; a[mi].h[7]  = (__bf16)f1.w;
            a[mi].h[8]  = (__bf16)f2.x; a[mi].h[9]  = (__bf16)f2.y;
            a[mi].h[10] = (__bf16)f2.z; a[mi].h[11] = (__bf16)f2.w;
            a[mi].h[12] = (__bf16)f3.x; a[mi].h[13] = (__bf16)f3.y;
            a[mi].h[14] = (__bf16)f3.z; a[mi].h[15] = (__bf16)f3.w;
        }

#pragma unroll
        for (int nt = 0; nt < 4; ++nt) {
            const __bf16* wrow = W + (size_t)(n0 + nt * 16 + row) * D_MODEL + kg + k0;
            Frag bf_;
            bf_.u[0] = *(const uint4*)(wrow);
            bf_.u[1] = *(const uint4*)(wrow + 16);
            acc[0][nt] = wmma_bf16(a[0].v, bf_.v, acc[0][nt]);
            acc[1][nt] = wmma_bf16(a[1].v, bf_.v, acc[1][nt]);
        }
    }

#pragma unroll
    for (int nt = 0; nt < 4; ++nt) {
        const int colg = n0 + nt * 16 + row;
        const float bb = bias[colg];
#pragma unroll
        for (int mi = 0; mi < 2; ++mi)
#pragma unroll
            for (int vv = 0; vv < 8; ++vv)
                O[(size_t)(m0 + mi * 16 + kg + vv) * D_MODEL + colg] =
                    (__bf16)(acc[mi][nt][vv] + bb);
    }
}

// ---------------------------------------------------------------------------
// Kernel 3: flash attention per (b, h, 16-row q tile), online softmax + relpos
// ---------------------------------------------------------------------------
__global__ __launch_bounds__(128) void flash_attn_kernel(
    const __bf16* __restrict__ qp, const __bf16* __restrict__ kp,
    const __bf16* __restrict__ vp, const float* __restrict__ relpos,
    __bf16* __restrict__ cx)
{
    __shared__ __bf16 lds_v[ATT_WAVES][32][72];  // V chunk, padded rows
    __shared__ float  lds_p[ATT_WAVES][16][34];  // probs C->A layout bounce

    const int w    = threadIdx.x >> 5;
    const int lane = threadIdx.x & 31;
    const int col  = lane & 15;
    const int kg   = (lane >> 4) << 3;           // 0 or 8
    const int b    = blockIdx.z;
    const int h    = blockIdx.y;
    const int qbase = (blockIdx.x * ATT_WAVES + w) * 16;
    const int hoff = h * DK;

    // LDS byte offset of this lane's V row (flat LDS aperture: offset = addr[31:0])
    const unsigned vrow_lds = (unsigned)(size_t)&lds_v[w][lane][0];

    // Q fragments (A-layout), d_k = 64 split into two K=32 halves
    Frag qa0, qa1;
    {
        const __bf16* qrow = qp + (size_t)(b * SEQ + qbase + col) * D_MODEL + hoff + kg;
        qa0.u[0] = *(const uint4*)(qrow);
        qa0.u[1] = *(const uint4*)(qrow + 16);
        qa1.u[0] = *(const uint4*)(qrow + 32);
        qa1.u[1] = *(const uint4*)(qrow + 48);
    }

    float mrow[8], lrow[8];
    v8f c[4];
#pragma unroll
    for (int i = 0; i < 8; ++i) { mrow[i] = -1.0e30f; lrow[i] = 0.0f; }
#pragma unroll
    for (int nt = 0; nt < 4; ++nt) c[nt] = zero8();

    const float scale = 0.125f;  // 1/sqrt(64)
    const float* rpbase = relpos + (size_t)(h * SEQ + qbase + kg) * SEQ;

    for (int kb = 0; kb < SEQ; kb += 32) {
        // async-stage V chunk (32 keys x 64 dims) into LDS, lane -> key kb+lane
        {
            const __bf16* vsrc = vp + (size_t)(b * SEQ + kb + lane) * D_MODEL + hoff;
#pragma unroll
            for (int j = 0; j < 8; ++j)
                async_g2l_16B(vrow_lds + (unsigned)(j * 16), vsrc + j * 8);
        }

        // scores: two 16x16 tiles (keys kb..kb+15 and kb+16..kb+31)
        v8f s0, s1;
        {
            const __bf16* krow0 = kp + (size_t)(b * SEQ + kb + col) * D_MODEL + hoff + kg;
            Frag f0, f1;
            f0.u[0] = *(const uint4*)(krow0);
            f0.u[1] = *(const uint4*)(krow0 + 16);
            f1.u[0] = *(const uint4*)(krow0 + 32);
            f1.u[1] = *(const uint4*)(krow0 + 48);
            v8f t = wmma_bf16(qa1.v, f1.v, zero8());
            s0 = wmma_bf16(qa0.v, f0.v, t);

            const __bf16* krow1 = krow0 + (size_t)16 * D_MODEL;
            f0.u[0] = *(const uint4*)(krow1);
            f0.u[1] = *(const uint4*)(krow1 + 16);
            f1.u[0] = *(const uint4*)(krow1 + 32);
            f1.u[1] = *(const uint4*)(krow1 + 48);
            t = wmma_bf16(qa1.v, f1.v, zero8());
            s1 = wmma_bf16(qa0.v, f0.v, t);
        }

        // scale + relpos bias, per-lane partial row max
        float sv0[8], sv1[8], cmax[8];
        const float* rp = rpbase + kb;
#pragma unroll
        for (int vv = 0; vv < 8; ++vv) {
            float r0 = rp[(size_t)vv * SEQ + col];
            float r1 = rp[(size_t)vv * SEQ + 16 + col];
            sv0[vv] = s0[vv] * scale + r0;
            sv1[vv] = s1[vv] * scale + r1;
            cmax[vv] = fmaxf(sv0[vv], sv1[vv]);
        }
        // row reduce across the 16-lane half (rows 0-7 in lanes 0-15, 8-15 in 16-31)
#pragma unroll
        for (int m = 1; m < 16; m <<= 1)
#pragma unroll
            for (int vv = 0; vv < 8; ++vv)
                cmax[vv] = fmaxf(cmax[vv], __shfl_xor(cmax[vv], m, 32));

        float csum[8];
#pragma unroll
        for (int vv = 0; vv < 8; ++vv) {
            float nm = fmaxf(mrow[vv], cmax[vv]);
            float rs = __expf(mrow[vv] - nm);
            sv0[vv] = __expf(sv0[vv] - nm);
            sv1[vv] = __expf(sv1[vv] - nm);
            csum[vv] = sv0[vv] + sv1[vv];
            mrow[vv] = nm;
            lrow[vv] *= rs;
#pragma unroll
            for (int nt = 0; nt < 4; ++nt) c[nt][vv] *= rs;
        }
#pragma unroll
        for (int m = 1; m < 16; m <<= 1)
#pragma unroll
            for (int vv = 0; vv < 8; ++vv)
                csum[vv] += __shfl_xor(csum[vv], m, 32);
#pragma unroll
        for (int vv = 0; vv < 8; ++vv) lrow[vv] += csum[vv];

        // bounce P through LDS: C-layout -> A-layout (bf16)
#pragma unroll
        for (int vv = 0; vv < 8; ++vv) {
            lds_p[w][kg + vv][col]      = sv0[vv];
            lds_p[w][kg + vv][16 + col] = sv1[vv];
        }
        wave_lds_sync();

        Frag pa;
#pragma unroll
        for (int j = 0; j < 8; ++j) {
            pa.h[j]     = (__bf16)lds_p[w][col][kg + j];
            pa.h[8 + j] = (__bf16)lds_p[w][col][kg + 16 + j];
        }

        // make sure the async V tile has landed before gathering B fragments
        wait_async0();

        // ctx += P(16x32) * V(32x64), 4 N-tiles of 16
#pragma unroll
        for (int nt = 0; nt < 4; ++nt) {
            Frag vb;
            const int d = nt * 16 + col;
#pragma unroll
            for (int j = 0; j < 8; ++j) {
                vb.h[j]     = lds_v[w][kg + j][d];
                vb.h[8 + j] = lds_v[w][kg + 16 + j][d];
            }
            c[nt] = wmma_bf16(pa.v, vb.v, c[nt]);
        }
        wave_lds_sync();  // drain LDS reads before next chunk's async stores
    }

    // normalize and store ctx (combined-heads layout [B,S,D_MODEL], bf16)
    float inv[8];
#pragma unroll
    for (int vv = 0; vv < 8; ++vv) inv[vv] = 1.0f / lrow[vv];
    __bf16* crow = cx + (size_t)(b * SEQ + qbase + kg) * D_MODEL + hoff;
#pragma unroll
    for (int nt = 0; nt < 4; ++nt)
#pragma unroll
        for (int vv = 0; vv < 8; ++vv)
            crow[(size_t)vv * D_MODEL + nt * 16 + col] = (__bf16)(c[nt][vv] * inv[vv]);
}

// ---------------------------------------------------------------------------
// Kernel 4: output projection  out = ctx * Wo^T + bo   (bf16 x bf16 -> fp32)
// wave -> 32x64 output tile
// ---------------------------------------------------------------------------
__global__ __launch_bounds__(256) void out_proj_kernel(
    const __bf16* __restrict__ cxin, const __bf16* __restrict__ Wo,
    const float* __restrict__ bo, float* __restrict__ out)
{
    const int wave = threadIdx.x >> 5;
    const int lane = threadIdx.x & 31;
    const int row  = lane & 15;
    const int kg   = (lane >> 4) << 3;
    const int m0   = blockIdx.x * 32;
    const int n0   = (blockIdx.y * 8 + wave) * 64;

    v8f acc[2][4];
#pragma unroll
    for (int mi = 0; mi < 2; ++mi)
#pragma unroll
        for (int nt = 0; nt < 4; ++nt) acc[mi][nt] = zero8();

    const __bf16* arow0 = cxin + (size_t)(m0 + row) * D_MODEL + kg;
    const __bf16* arow1 = arow0 + (size_t)16 * D_MODEL;

    for (int k0 = 0; k0 < D_MODEL; k0 += 32) {
        Frag a0, a1;
        a0.u[0] = *(const uint4*)(arow0 + k0);
        a0.u[1] = *(const uint4*)(arow0 + k0 + 16);
        a1.u[0] = *(const uint4*)(arow1 + k0);
        a1.u[1] = *(const uint4*)(arow1 + k0 + 16);
#pragma unroll
        for (int nt = 0; nt < 4; ++nt) {
            const __bf16* wrow = Wo + (size_t)(n0 + nt * 16 + row) * D_MODEL + kg + k0;
            Frag bf_;
            bf_.u[0] = *(const uint4*)(wrow);
            bf_.u[1] = *(const uint4*)(wrow + 16);
            acc[0][nt] = wmma_bf16(a0.v, bf_.v, acc[0][nt]);
            acc[1][nt] = wmma_bf16(a1.v, bf_.v, acc[1][nt]);
        }
    }

#pragma unroll
    for (int nt = 0; nt < 4; ++nt) {
        const int colg = n0 + nt * 16 + row;
        const float bb = bo[colg];
#pragma unroll
        for (int mi = 0; mi < 2; ++mi)
#pragma unroll
            for (int vv = 0; vv < 8; ++vv)
                out[(size_t)(m0 + mi * 16 + kg + vv) * D_MODEL + colg] =
                    acc[mi][nt][vv] + bb;
    }
}

// ---------------------------------------------------------------------------
extern "C" void kernel_launch(void* const* d_in, const int* in_sizes, int n_in,
                              void* d_out, int out_size, void* d_ws, size_t ws_size,
                              hipStream_t stream) {
    const float* Q      = (const float*)d_in[0];
    const float* Kin    = (const float*)d_in[1];
    const float* V      = (const float*)d_in[2];
    const float* Wq     = (const float*)d_in[3];
    const float* bq     = (const float*)d_in[4];
    const float* Wk     = (const float*)d_in[5];
    const float* bk     = (const float*)d_in[6];
    const float* Wv     = (const float*)d_in[7];
    const float* bv     = (const float*)d_in[8];
    const float* Wo     = (const float*)d_in[9];
    const float* bo     = (const float*)d_in[10];
    const float* relpos = (const float*)d_in[11];

    __bf16* ws = (__bf16*)d_ws;
    const size_t WSZ = (size_t)D_MODEL * D_MODEL;        // 1,048,576 elems
    const size_t PSZ = (size_t)BATCH * SEQ * D_MODEL;    // 8,388,608 elems
    __bf16* wq_b = ws;
    __bf16* wk_b = ws + WSZ;
    __bf16* wv_b = ws + 2 * WSZ;
    __bf16* wo_b = ws + 3 * WSZ;
    __bf16* qp   = ws + 4 * WSZ;
    __bf16* kp   = qp + PSZ;
    __bf16* vp   = kp + PSZ;
    __bf16* cx   = vp + PSZ;

    // 1) weight casts
    {
        dim3 grid((unsigned)(WSZ / 4 / 256)), block(256);
        cast_w_kernel<<<grid, block, 0, stream>>>(Wq, wq_b, (int)WSZ);
        cast_w_kernel<<<grid, block, 0, stream>>>(Wk, wk_b, (int)WSZ);
        cast_w_kernel<<<grid, block, 0, stream>>>(Wv, wv_b, (int)WSZ);
        cast_w_kernel<<<grid, block, 0, stream>>>(Wo, wo_b, (int)WSZ);
    }

    // 2) q/k/v projections
    {
        dim3 grid(BATCH * SEQ / 32, D_MODEL / (64 * 8), 3), block(256);
        proj_qkv_kernel<<<grid, block, 0, stream>>>(Q, Kin, V, wq_b, wk_b, wv_b,
                                                    bq, bk, bv, qp, kp, vp);
    }

    // 3) flash attention
    {
        dim3 grid(SEQ / 16 / ATT_WAVES, NHEAD, BATCH), block(32 * ATT_WAVES);
        flash_attn_kernel<<<grid, block, 0, stream>>>(qp, kp, vp, relpos, cx);
    }

    // 4) output projection
    {
        dim3 grid(BATCH * SEQ / 32, D_MODEL / (64 * 8), 1), block(256);
        out_proj_kernel<<<grid, block, 0, stream>>>(cx, wo_b, bo, (float*)d_out);
    }
}
